// FluidLayer_66194035966266
// MI455X (gfx1250) — compile-verified, matrix operand order checked by reading, not run on hardware
//
#include <hip/hip_runtime.h>
#include <hip/hip_bf16.h>
#include <math.h>

typedef __attribute__((ext_vector_type(16))) _Float16 v16h;
typedef __attribute__((ext_vector_type(8)))  float    v8f;

#define DD   1024
#define SS   2048
#define MM   8192   // B*S = 4*2048

// ---------------------------------------------------------------------------
// f32 -> f16 weight conversion WITH transpose: dst[n*K + k] = src[k*N + n]
// (so GEMM B-staging is contiguous along K, enabling b128 async copies)
// ---------------------------------------------------------------------------
__global__ __launch_bounds__(256) void conv_f2h_T(const float* __restrict__ src,
                                                  _Float16* __restrict__ dst,
                                                  int K, int N) {
    int i = blockIdx.x * 256 + threadIdx.x;
    if (i < K * N) {
        int k = i / N, n = i % N;                  // coalesced read
        dst[(size_t)n * K + k] = (_Float16)src[i];
    }
}

// init: xa = x, xh = f16(x), acc = 0
__global__ __launch_bounds__(256) void init_state(const float* __restrict__ x,
                                                  float* __restrict__ xa,
                                                  _Float16* __restrict__ xh,
                                                  float* __restrict__ acc, int n) {
    int i = blockIdx.x * 256 + threadIdx.x;
    if (i < n) {
        float v = x[i];
        xa[i]  = v;
        xh[i]  = (_Float16)v;
        acc[i] = 0.0f;
    }
}

// ---------------------------------------------------------------------------
// Async global->LDS staging of a 128x32-half tile (rows contiguous along K).
// Each chunk is 16B; GLOBAL_LOAD_ASYNC_TO_LDS_B128 moves 16B/lane, tracked by
// ASYNCcnt, no VGPR round-trip.  512 chunks / 256 threads = 2 per thread.
// ---------------------------------------------------------------------------
__device__ inline void stage_async(const _Float16* __restrict__ gsrc,  // tile origin
                                   int gstride,                        // row stride (halves)
                                   _Float16* lds, int tid) {
#pragma unroll
    for (int q = tid; q < 512; q += 256) {
        int row = q >> 2, qc = q & 3;
        const _Float16* g = gsrc + (size_t)row * gstride + qc * 8;
        uint32_t loff = (uint32_t)(uintptr_t)(lds + row * 32 + qc * 8);
        asm volatile("global_load_async_to_lds_b128 %0, %1, off"
                     :: "v"(loff), "v"((uint64_t)(uintptr_t)g)
                     : "memory");
    }
}

__device__ inline void wait_async0() {
    asm volatile("s_wait_asynccnt 0x0" ::: "memory");
}

// ---------------------------------------------------------------------------
// WMMA fragment load from LDS tile (16 rows x 32 halves, row stride 32).
// 16-bit A-layout (ISA 7.12.2): lane<16 -> K words {0..3}+{8..11};
// lane>=16 -> words {4..7}+{12..15}.  Both groups 16B-contiguous => 2x b128.
// ---------------------------------------------------------------------------
__device__ inline v16h frag_from_lds(const _Float16* base, int rowOff, int lane) {
    const uint4* row = (const uint4*)(base + (size_t)(rowOff + (lane & 15)) * 32);
    int q = (lane < 16) ? 0 : 1;
    union { v16h v; uint4 q4[2]; } u;
    u.q4[0] = row[q];
    u.q4[1] = row[q + 2];
    return u.v;
}

// ---------------------------------------------------------------------------
// Tiled f16 WMMA GEMM:  out[M,N] = epi(A[M,K] @ Bt[N,K]^T + bias[N])
// Block = 256 thr (8 waves), block tile 128x128, wave tile 64x32 (4x2 WMMA),
// K staged 32-wide, double-buffered LDS filled by async loads.
// EPI: 0 = identity -> f32, 1 = exact GELU -> f16, 2 = sigmoid -> f16.
// ---------------------------------------------------------------------------
template <int EPI>
__global__ __launch_bounds__(256) void gemm_wmma(const _Float16* __restrict__ A,
                                                 const _Float16* __restrict__ Bt,
                                                 const float* __restrict__ bias,
                                                 void* __restrict__ out,
                                                 int M, int N, int K) {
    __shared__ _Float16 sA[2][128 * 32];   // [m][k] tiles
    __shared__ _Float16 sB[2][128 * 32];   // [n][k] tiles
    const int tid  = threadIdx.x;
    const int lane = tid & 31;
    const int wave = tid >> 5;
    const int wm   = wave >> 2;            // 0..1  (64 rows each)
    const int wn   = wave & 3;             // 0..3  (32 cols each)
    const int m0   = blockIdx.y * 128;
    const int n0   = blockIdx.x * 128;

    const _Float16* Ab = A  + (size_t)m0 * K;
    const _Float16* Bb = Bt + (size_t)n0 * K;

    v8f acc[4][2];
#pragma unroll
    for (int i = 0; i < 4; ++i)
#pragma unroll
        for (int j = 0; j < 2; ++j) acc[i][j] = (v8f){0,0,0,0,0,0,0,0};

    // prologue: stage k-slice 0 into buffer 0
    stage_async(Ab, K, &sA[0][0], tid);
    stage_async(Bb, K, &sB[0][0], tid);
    wait_async0();
    __syncthreads();

    int cbuf = 0;
    for (int k0 = 32; k0 <= K; k0 += 32) {
        if (k0 < K) {   // stage next slice while computing current
            stage_async(Ab + k0, K, &sA[cbuf ^ 1][0], tid);
            stage_async(Bb + k0, K, &sB[cbuf ^ 1][0], tid);
        }
        v16h af[4], bf[2];
#pragma unroll
        for (int i = 0; i < 4; ++i) af[i] = frag_from_lds(&sA[cbuf][0], wm * 64 + i * 16, lane);
#pragma unroll
        for (int j = 0; j < 2; ++j) bf[j] = frag_from_lds(&sB[cbuf][0], wn * 32 + j * 16, lane);
#pragma unroll
        for (int i = 0; i < 4; ++i)
#pragma unroll
            for (int j = 0; j < 2; ++j)
                acc[i][j] = __builtin_amdgcn_wmma_f32_16x16x32_f16(
                    false, af[i], false, bf[j], (short)0, acc[i][j], false, false);
        if (k0 < K) {
            wait_async0();
            __syncthreads();
            cbuf ^= 1;
        }
    }

    // epilogue: C/D layout -> row = v + 8*(lane>=16), col = lane&15
    const int colBase   = n0 + wn * 32 + (lane & 15);
    const int rowInLane = (lane < 16) ? 0 : 8;
#pragma unroll
    for (int i = 0; i < 4; ++i)
#pragma unroll
        for (int j = 0; j < 2; ++j) {
            int col = colBase + j * 16;
            float bv = bias[col];
#pragma unroll
            for (int v = 0; v < 8; ++v) {
                int row = m0 + wm * 64 + i * 16 + rowInLane + v;
                float x = acc[i][j][v] + bv;
                size_t idx = (size_t)row * N + col;
                if (EPI == 0) {
                    ((float*)out)[idx] = x;
                } else if (EPI == 1) {  // exact GELU
                    float g = 0.5f * x * (1.0f + erff(x * 0.70710678118654752f));
                    ((_Float16*)out)[idx] = (_Float16)g;
                } else {                // sigmoid
                    float s = 1.0f / (1.0f + expf(-x));
                    ((_Float16*)out)[idx] = (_Float16)s;
                }
            }
        }
}

// ---------------------------------------------------------------------------
// Fused: multi-dilation causal Laplacian + gated acc + Euler + LayerNorm.
// One block per (b,t) row; 256 threads x 4 channels. Writes f32 + f16 x_next.
// ---------------------------------------------------------------------------
__global__ __launch_bounds__(256) void update_ln(const float* __restrict__ xc,
                                                 const float* __restrict__ react,
                                                 const _Float16* __restrict__ gate,
                                                 float* __restrict__ acc,
                                                 const float* __restrict__ dc,
                                                 const float* __restrict__ gamma,
                                                 const float* __restrict__ beta,
                                                 const float* __restrict__ dtp,
                                                 float* __restrict__ xn,
                                                 _Float16* __restrict__ xh) {
    const int r = blockIdx.x;            // 0..MM-1
    const int t = r & (SS - 1);          // position in sequence
    const int tid = threadIdx.x;
    const float dt = *dtp;
    const size_t base = (size_t)r * DD;

    __shared__ float red[256];
    float y[4];
    float sum = 0.0f, sumsq = 0.0f;

#pragma unroll
    for (int e = 0; e < 4; ++e) {
        int c = tid + e * 256;
        float xv = xc[base + c];
        float diff = 0.0f;
        const int dil[3] = {1, 4, 16};
#pragma unroll
        for (int i = 0; i < 3; ++i) {
            int d = dil[i];
            float xm1 = (t >= d)     ? xc[base - (size_t)d * DD + c]     : 0.0f;
            float xm2 = (t >= 2 * d) ? xc[base - (size_t)2 * d * DD + c] : 0.0f;
            diff += (xm2 - 2.0f * xm1 + xv) * dc[i * DD + c];
        }
        float rv = react[base + c];
        float g  = (float)gate[base + c];            // sigmoid already applied
        float a  = acc[base + c] + g * tanhf(rv);
        acc[base + c] = a;
        float yv = xv + dt * (diff + rv + 0.05f * a);
        y[e] = yv;
        sum += yv;
        sumsq += yv * yv;
    }

    red[tid] = sum; __syncthreads();
    for (int s2 = 128; s2 > 0; s2 >>= 1) {
        if (tid < s2) red[tid] += red[tid + s2];
        __syncthreads();
    }
    float mu = red[0] * (1.0f / DD);
    __syncthreads();
    red[tid] = sumsq; __syncthreads();
    for (int s2 = 128; s2 > 0; s2 >>= 1) {
        if (tid < s2) red[tid] += red[tid + s2];
        __syncthreads();
    }
    float var = red[0] * (1.0f / DD) - mu * mu;
    float inv = rsqrtf(var + 1e-5f);

#pragma unroll
    for (int e = 0; e < 4; ++e) {
        int c = tid + e * 256;
        float o = (y[e] - mu) * inv * gamma[c] + beta[c];
        xn[base + c] = o;
        xh[base + c] = (_Float16)o;
    }
}

// ---------------------------------------------------------------------------
// Host launcher
// ---------------------------------------------------------------------------
extern "C" void kernel_launch(void* const* d_in, const int* in_sizes, int n_in,
                              void* d_out, int out_size, void* d_ws, size_t ws_size,
                              hipStream_t stream) {
    const float* x  = (const float*)d_in[0];
    const float* W1 = (const float*)d_in[1];
    const float* b1 = (const float*)d_in[2];
    const float* W2 = (const float*)d_in[3];
    const float* b2 = (const float*)d_in[4];
    const float* Wg = (const float*)d_in[5];
    const float* bg = (const float*)d_in[6];
    const float* dc = (const float*)d_in[7];
    const float* gamma = (const float*)d_in[8];
    const float* beta  = (const float*)d_in[9];
    const float* dtp   = (const float*)d_in[10];
    // d_in[11] = max_steps (fixed at 8 per setup)

    const size_t nX = (size_t)MM * DD;        // 8,388,608 elems

    char* w = (char*)d_ws;
    float*    xa    = (float*)w;            w += nX * 4;                // 32 MB
    float*    xb    = (float*)w;            w += nX * 4;                // 32 MB
    _Float16* xh    = (_Float16*)w;         w += nX * 2;                // 16 MB
    _Float16* Hh    = (_Float16*)w;         w += (size_t)MM * 2048 * 2; // 32 MB
    _Float16* gh    = (_Float16*)w;         w += nX * 2;                // 16 MB
    float*    react = (float*)w;            w += nX * 4;                // 32 MB
    float*    accB  = (float*)w;            w += nX * 4;                // 32 MB
    _Float16* W1t   = (_Float16*)w;         w += (size_t)DD * 2048 * 2; // [2048,1024]
    _Float16* W2t   = (_Float16*)w;         w += (size_t)2048 * DD * 2; // [1024,2048]
    _Float16* Wgt   = (_Float16*)w;         w += (size_t)DD * DD * 2;   // [1024,1024]

    // transpose+convert weights (runs once per launch; cheap)
    conv_f2h_T<<<(2097152 + 255) / 256, 256, 0, stream>>>(W1, W1t, DD,   2048);
    conv_f2h_T<<<(2097152 + 255) / 256, 256, 0, stream>>>(W2, W2t, 2048, DD);
    conv_f2h_T<<<(1048576 + 255) / 256, 256, 0, stream>>>(Wg, Wgt, DD,   DD);

    init_state<<<(int)(nX / 256), 256, 0, stream>>>(x, xa, xh, accB, (int)nX);

    for (int s = 0; s < 8; ++s) {
        const float* xcur = (s & 1) ? xb : xa;
        float* xnxt = (s == 7) ? (float*)d_out : ((s & 1) ? xa : xb);

        // H = GELU(x @ W1 + b1)          -> f16
        gemm_wmma<1><<<dim3(16, 64), 256, 0, stream>>>(xh, W1t, b1, Hh, MM, 2048, DD);
        // gate = sigmoid(x @ Wg + bg)    -> f16
        gemm_wmma<2><<<dim3(8, 64), 256, 0, stream>>>(xh, Wgt, bg, gh, MM, DD, DD);
        // react = H @ W2 + b2            -> f32
        gemm_wmma<0><<<dim3(8, 64), 256, 0, stream>>>(Hh, W2t, b2, react, MM, DD, 2048);
        // Laplacian + acc + Euler + LayerNorm -> x_next (f32 + f16 shadow)
        update_ln<<<MM, 256, 0, stream>>>(xcur, react, gh, accB, dc, gamma, beta,
                                          dtp, xnxt, xh);
    }
}
